// AirGNN_82274393522970
// MI455X (gfx1250) — matrix operand order repaired and checked by reading, not compiled, not run on hardware
//
#include <hip/hip_runtime.h>

// ---------------- problem constants (match reference) ----------------
constexpr int kFIN  = 602;   // input features
constexpr int kHID  = 256;   // hidden
constexpr int kC    = 47;    // classes
constexpr int kCP   = 48;    // padded classes (16-aligned for WMMA tiles)
constexpr int kKC   = 32;    // WMMA K chunk (bf16 16x16x32)
constexpr int kNKC  = (kFIN + kKC - 1) / kKC;  // 19 chunks over 602
constexpr float kLAM = 0.5f; // gamma * LAMBDA_AMP = 0.5 (lambda=0.5 -> gamma=1)

// ---------------- WMMA types ----------------
typedef __attribute__((ext_vector_type(16))) __bf16 v16bf;
typedef __attribute__((ext_vector_type(8)))  float  v8f;

union Frag16 { v16bf v; unsigned int u[8]; };
union Acc8   { v8f   v; float        f[8]; };

// truncating f32 -> bf16 (1 VALU); error < one bf16 mantissa step
__device__ __forceinline__ unsigned short f2bf(float f) {
  return (unsigned short)(__float_as_uint(f) >> 16);
}
// pack two f32 into packed bf16 pair: single v_perm_b32-class op
__device__ __forceinline__ unsigned int pack2bf(float lo, float hi) {
  return (__float_as_uint(hi) & 0xFFFF0000u) | (__float_as_uint(lo) >> 16);
}

// ---------------- graph normalization ----------------
__global__ void k_deg_init(float* deg, int N) {
  int i = blockIdx.x * blockDim.x + threadIdx.x;
  if (i < N) deg[i] = 1.0f;                 // self loop
}
__global__ void k_deg_count(const int* __restrict__ row, float* deg, int E) {
  int e = blockIdx.x * blockDim.x + threadIdx.x;
  if (e < E) atomicAdd(&deg[row[e]], 1.0f);
}
__global__ void k_norm(const float* __restrict__ deg, float* dinv, float* selfw, int N) {
  int i = blockIdx.x * blockDim.x + threadIdx.x;
  if (i < N) { float d = deg[i]; dinv[i] = rsqrtf(d); selfw[i] = 1.0f / d; }
}
__global__ void k_edge_w(const int* __restrict__ row, const int* __restrict__ col,
                         const float* __restrict__ dinv, float* w, int E) {
  int e = blockIdx.x * blockDim.x + threadIdx.x;
  if (e < E) w[e] = dinv[row[e]] * dinv[col[e]];
}

// ---------------- fused MLP: hh = relu(x@W1+b1)@W2+b2 (bf16 WMMA) ----------------
// Block: 256 threads = 8 wave32. BM=32 rows, BN=256 hidden cols.
// Phase 1: 19 K-chunks of 32 from x/W1 via LDS -> 4 WMMA/wave/chunk.
//   18 full chunks: branchless b64/b32 loads + packed ds_store_b32.
//   1 tail chunk (26 valid K): guarded per element.
// Phase 2: hidden tile (LDS bf16) @ W2 (LDS bf16, col-padded to 48).
__global__ void __launch_bounds__(256)
k_mlp(const float* __restrict__ x, const float* __restrict__ W1,
      const float* __restrict__ b1, const float* __restrict__ W2,
      const float* __restrict__ b2, float* __restrict__ hh,
      float* __restrict__ h, int N)
{
  __shared__ alignas(16) unsigned short As  [32][40];    // x chunk  (rows x K), bf16
  __shared__ alignas(16) unsigned short Bts [256][40];   // W1 chunk transposed (col x K)
  __shared__ alignas(16) unsigned short H1s [32][264];   // hidden tile (rows x 256)
  __shared__ alignas(16) unsigned short W2ts[48][264];   // W2 transposed (col x 256)

  const int t    = threadIdx.x;
  const int wave = t >> 5;
  const int lane = t & 31;
  const int l16  = lane & 15;
  const int hiA  = (lane >= 16) ? 8  : 0;   // A-fragment K offset (ISA layout)
  const int hiB  = (lane >= 16) ? 8  : 0;   // B-fragment K offset (ISA layout)
  const long long rowBase = (long long)blockIdx.x * 32;

  // staging coordinates for the x tile
  const int sr  = t >> 3;            // LDS row 0..31
  const int sc4 = (t & 7) * 4;       // K sub-offset 0..28 step 4
  long long gr = rowBase + sr;
  if (gr > (long long)N - 1) gr = (long long)N - 1;   // clamp, never branch
  const float* xrow = x + gr * kFIN;

  // stage W2 -> LDS bf16 transposed (packed pairs), pad column 47 with zeros
  for (int idx = t; idx < kCP * (kHID / 2); idx += 256) {
    int n = idx / (kHID / 2);
    int kp = (idx - n * (kHID / 2)) * 2;
    float va = (n < kC) ? W2[kp * kC + n]       : 0.0f;
    float vb = (n < kC) ? W2[(kp + 1) * kC + n] : 0.0f;
    *(unsigned int*)&W2ts[n][kp] = pack2bf(va, vb);
  }

  Acc8 acc[2][2];   // [row tile][col tile in {wave, wave+8}]
  for (int a = 0; a < 2; ++a)
    for (int b = 0; b < 2; ++b)
      for (int i = 0; i < 8; ++i) acc[a][b].f[i] = 0.0f;

  for (int kc = 0; kc < kNKC; ++kc) {
    const int k0 = kc * kKC;
    __syncthreads();
    if (k0 + kKC <= kFIN) {
      // ---- full chunk: branchless staging ----
      const float* xp = xrow + k0 + sc4;
      if (kc + 1 < kNKC) __builtin_prefetch(xp + kKC, 0, 1);  // global_prefetch_b8
      float2 v01 = *(const float2*)xp;          // 8B-aligned for any row
      float2 v23 = *(const float2*)(xp + 2);
      *(unsigned int*)&As[sr][sc4]     = pack2bf(v01.x, v01.y);
      *(unsigned int*)&As[sr][sc4 + 2] = pack2bf(v23.x, v23.y);
      // W1 chunk transposed: packed K-pairs, coalesced across t
#pragma unroll
      for (int i = 0; i < kKC; i += 2) {
        float va = W1[(long long)(k0 + i)     * kHID + t];
        float vb = W1[(long long)(k0 + i + 1) * kHID + t];
        *(unsigned int*)&Bts[t][i] = pack2bf(va, vb);
      }
    } else {
      // ---- tail chunk (26 valid K): guarded ----
#pragma unroll
      for (int j = 0; j < 4; j += 2) {
        int kk = k0 + sc4 + j;
        float va = (kk     < kFIN) ? xrow[kk]     : 0.0f;
        float vb = (kk + 1 < kFIN) ? xrow[kk + 1] : 0.0f;
        *(unsigned int*)&As[sr][sc4 + j] = pack2bf(va, vb);
      }
#pragma unroll
      for (int i = 0; i < kKC; i += 2) {
        int kk = k0 + i;
        float va = (kk     < kFIN) ? W1[(long long)kk       * kHID + t] : 0.0f;
        float vb = (kk + 1 < kFIN) ? W1[(long long)(kk + 1) * kHID + t] : 0.0f;
        *(unsigned int*)&Bts[t][i] = pack2bf(va, vb);
      }
    }
    __syncthreads();

    // ---- A fragments for both row tiles ----
    Frag16 a0, a1;
#pragma unroll
    for (int i = 0; i < 8; ++i) {
      int koff = hiA + ((i < 4) ? (2 * i) : (16 + 2 * (i - 4)));
      a0.u[i] = *(const unsigned int*)&As[l16][koff];
      a1.u[i] = *(const unsigned int*)&As[16 + l16][koff];
    }
    // ---- B fragments + WMMA for 2 column tiles per wave ----
#pragma unroll
    for (int cti = 0; cti < 2; ++cti) {
      int ct = wave + 8 * cti;
      Frag16 b;
#pragma unroll
      for (int i = 0; i < 8; ++i)
        b.u[i] = *(const unsigned int*)&Bts[16 * ct + l16][hiB + ((i < 4) ? (2 * i) : (16 + 2 * (i - 4)))];
      acc[0][cti].v = __builtin_amdgcn_wmma_f32_16x16x32_bf16(
          false, a0.v, false, b.v, (short)0, acc[0][cti].v, false, false);
      acc[1][cti].v = __builtin_amdgcn_wmma_f32_16x16x32_bf16(
          false, a1.v, false, b.v, (short)0, acc[1][cti].v, false, false);
    }
  }

  __syncthreads();
  // ---- bias + ReLU -> hidden tile in LDS (bf16) ----
#pragma unroll
  for (int cti = 0; cti < 2; ++cti) {
    int ct  = wave + 8 * cti;
    int col = 16 * ct + l16;
    float bias = b1[col];
#pragma unroll
    for (int rt = 0; rt < 2; ++rt) {
#pragma unroll
      for (int v = 0; v < 8; ++v) {
        int r = 16 * rt + ((lane < 16) ? v : v + 8);
        float val = acc[rt][cti].f[v] + bias;
        H1s[r][col] = f2bf(val > 0.0f ? val : 0.0f);
      }
    }
  }
  __syncthreads();

  // ---- phase 2: 32x48 output = H1s (32x256) @ W2ts^T, 6 tiles on waves 0..5 ----
  if (wave < 6) {
    int rt = wave / 3, ct = wave - rt * 3;
    Acc8 c2;
    for (int i = 0; i < 8; ++i) c2.f[i] = 0.0f;
#pragma unroll
    for (int kk = 0; kk < kHID / kKC; ++kk) {
      int kb = kk * kKC;
      Frag16 a, b;
#pragma unroll
      for (int i = 0; i < 8; ++i) {
        int koffA = kb + hiA + ((i < 4) ? (2 * i) : (16 + 2 * (i - 4)));
        a.u[i] = *(const unsigned int*)&H1s[16 * rt + l16][koffA];
        b.u[i] = *(const unsigned int*)&W2ts[16 * ct + l16][kb + hiB + ((i < 4) ? (2 * i) : (16 + 2 * (i - 4)))];
      }
      c2.v = __builtin_amdgcn_wmma_f32_16x16x32_bf16(
          false, a.v, false, b.v, (short)0, c2.v, false, false);
    }
    int col = 16 * ct + l16;
    float bias = (col < kC) ? b2[col] : 0.0f;
#pragma unroll
    for (int v = 0; v < 8; ++v) {
      long long r = rowBase + 16 * rt + ((lane < 16) ? v : v + 8);
      if (r < N) {
        float val = (col < kC) ? (c2.f[v] + bias) : 0.0f;  // zero padding col
        hh[r * kCP + col] = val;
        h [r * kCP + col] = val;   // h0 = hh
      }
    }
  }
}

// ---------------- AMP propagation step ----------------
// p = selfw[n] * h[n]  (init; padding stays 0 because h pad == 0), float4-wide
__global__ void k_prop_init(const float* __restrict__ h, const float* __restrict__ selfw,
                            float* __restrict__ p, int N) {
  int i = blockIdx.x * blockDim.x + threadIdx.x;   // float4 index
  int total = N * (kCP / 4);                       // N * 12
  if (i < total) {
    int n = i / (kCP / 4);
    float s = selfw[n];
    float4 v = ((const float4*)h)[i];
    v.x *= s; v.y *= s; v.z *= s; v.w *= s;
    ((float4*)p)[i] = v;
  }
}
// one wave32 per edge: lane handles channels {lane, lane+32} (<47); L2-resident atomics
__global__ void k_prop_edges(const int* __restrict__ row, const int* __restrict__ col,
                             const float* __restrict__ w, const float* __restrict__ h,
                             float* __restrict__ p, int E) {
  int gid  = blockIdx.x * blockDim.x + threadIdx.x;
  int e    = gid >> 5;
  int lane = gid & 31;
  if (e >= E) return;
  int r = row[e], c = col[e];
  float we = w[e];
  const float* hc = h + (long long)c * kCP;
  float*       pr = p + (long long)r * kCP;
  atomicAdd(&pr[lane], we * hc[lane]);
  if (lane < kC - 32) atomicAdd(&pr[lane + 32], we * hc[lane + 32]);
}
// proximal L21 update: z = p - hh; h = hh + max(||z||-lam,0)/||z|| * z
__global__ void k_prox(const float* __restrict__ p, const float* __restrict__ hh,
                       float* __restrict__ h, int N) {
  int gid  = blockIdx.x * blockDim.x + threadIdx.x;
  int n    = gid >> 5;
  int lane = gid & 31;
  if (n >= N) return;
  const float* pr = p  + (long long)n * kCP;
  const float* hr = hh + (long long)n * kCP;
  float z0 = pr[lane] - hr[lane];
  float z1 = (lane < 16) ? (pr[lane + 32] - hr[lane + 32]) : 0.0f;  // pad col -> 0
  float ss = z0 * z0 + z1 * z1;
#pragma unroll
  for (int m = 16; m >= 1; m >>= 1) ss += __shfl_xor(ss, m, 32);
  float rn = sqrtf(ss);
  float score = 0.0f;
  if (rn > 0.0f) {
    float tt = rn - kLAM;
    score = (tt > 0.0f ? tt : 0.0f) / rn;
  }
  float* ho = h + (long long)n * kCP;
  ho[lane] = hr[lane] + score * z0;
  if (lane < 16) ho[lane + 32] = hr[lane + 32] + score * z1;
}

// ---------------- log-softmax over 47 classes (wave per node) ----------------
__global__ void k_lsm(const float* __restrict__ h, float* __restrict__ out, int N) {
  int gid  = blockIdx.x * blockDim.x + threadIdx.x;
  int n    = gid >> 5;
  int lane = gid & 31;
  if (n >= N) return;
  const float* hr = h + (long long)n * kCP;
  bool has1 = (lane + 32) < kC;
  float v0 = hr[lane];
  float v1 = has1 ? hr[lane + 32] : -3.402823466e38f;
  float m = fmaxf(v0, v1);
#pragma unroll
  for (int s = 16; s >= 1; s >>= 1) m = fmaxf(m, __shfl_xor(m, s, 32));
  float sum = __expf(v0 - m) + (has1 ? __expf(v1 - m) : 0.0f);
#pragma unroll
  for (int s = 16; s >= 1; s >>= 1) sum += __shfl_xor(sum, s, 32);
  float lse = m + __logf(sum);
  out[(long long)n * kC + lane] = v0 - lse;
  if (has1) out[(long long)n * kC + lane + 32] = v1 - lse;
}

// ---------------- host launcher ----------------
extern "C" void kernel_launch(void* const* d_in, const int* in_sizes, int n_in,
                              void* d_out, int out_size, void* d_ws, size_t ws_size,
                              hipStream_t stream) {
  const float* x  = (const float*)d_in[0];
  const int*   row = (const int*)d_in[1];
  const int*   col = (const int*)d_in[2];
  const float* W1 = (const float*)d_in[3];
  const float* b1 = (const float*)d_in[4];
  const float* W2 = (const float*)d_in[5];
  const float* b2 = (const float*)d_in[6];
  float* out = (float*)d_out;

  const int E = in_sizes[1];
  const int N = in_sizes[0] / kFIN;

  // scratch carve-out (256B aligned)
  char* ws = (char*)d_ws;
  size_t off = 0;
  auto carve = [&](size_t bytes) -> void* {
    void* ptr = ws + off;
    off = (off + bytes + 255) & ~((size_t)255);
    return ptr;
  };
  float* deg   = (float*)carve((size_t)N * 4);
  float* dinv  = (float*)carve((size_t)N * 4);
  float* selfw = (float*)carve((size_t)N * 4);
  float* w     = (float*)carve((size_t)E * 4);
  float* hh    = (float*)carve((size_t)N * kCP * 4);
  float* h     = (float*)carve((size_t)N * kCP * 4);
  float* p     = (float*)carve((size_t)N * kCP * 4);
  (void)ws_size; (void)n_in; (void)out_size;

  const int B = 256;
  // normalization
  k_deg_init <<<(N + B - 1) / B, B, 0, stream>>>(deg, N);
  k_deg_count<<<(E + B - 1) / B, B, 0, stream>>>(row, deg, E);
  k_norm     <<<(N + B - 1) / B, B, 0, stream>>>(deg, dinv, selfw, N);
  k_edge_w   <<<(E + B - 1) / B, B, 0, stream>>>(row, col, dinv, w, E);

  // fused MLP (WMMA): hh = relu(x@W1+b1)@W2+b2 ; h = hh
  k_mlp<<<(N + 31) / 32, B, 0, stream>>>(x, W1, b1, W2, b2, hh, h, N);

  // K = 10 AMP iterations (y == propagate(h) since lambda = 0.5)
  for (int it = 0; it < 10; ++it) {
    k_prop_init <<<((size_t)N * (kCP / 4) + B - 1) / B, B, 0, stream>>>(h, selfw, p, N);
    k_prop_edges<<<((size_t)E * 32 + B - 1) / B, B, 0, stream>>>(row, col, w, h, p, E);
    k_prox      <<<((size_t)N * 32 + B - 1) / B, B, 0, stream>>>(p, hh, h, N);
  }

  // log-softmax -> output [N, 47]
  k_lsm<<<((size_t)N * 32 + B - 1) / B, B, 0, stream>>>(h, out, N);
}
